// MambaG2G_79345225827017
// MI455X (gfx1250) — compile-verified, matrix operand order checked by reading, not run on hardware
//
#include <hip/hip_runtime.h>
#include <hip/hip_bf16.h>
#include <math.h>

#define B_    8
#define L_    1024
#define DM    768
#define DS    16
#define DC    4
#define DI    1536
#define DTR   48
#define DTRP  64          // dt rank padded to WMMA-friendly K
#define DIMIN 768
#define DIMOUT 128
#define NROW  (B_ * L_)   // 8192

typedef __attribute__((ext_vector_type(16))) __bf16 v16bf;
typedef __attribute__((ext_vector_type(8)))  __bf16 v8bf;
typedef __attribute__((ext_vector_type(8)))  float  v8f;

__device__ __forceinline__ float sigmoidf_(float x) { return 1.0f / (1.0f + __expf(-x)); }
__device__ __forceinline__ float siluf_(float x)    { return x * sigmoidf_(x); }

// CDNA5 async global->LDS copy (ASYNCcnt-tracked, bypasses VGPRs).
// ldsoff is a raw LDS byte offset (dynamic-LDS base == 0 when kernel has no static LDS).
__device__ __forceinline__ void gl_async_b128(unsigned ldsoff, const void* g) {
    unsigned long long ga = (unsigned long long)g;
    asm volatile("global_load_async_to_lds_b128 %0, %1, off"
                 :: "v"(ldsoff), "v"(ga) : "memory");
}
__device__ __forceinline__ void wait_async0() {
    asm volatile("s_wait_asynccnt 0x0" ::: "memory");
}

// ---------------- RMSNorm (fp32 in -> bf16 out) ----------------
__global__ void __launch_bounds__(256)
k_rmsnorm(const float* __restrict__ x, const float* __restrict__ nw, __bf16* __restrict__ out) {
    int row = blockIdx.x;
    int tid = threadIdx.x;
    const float* xr = x + (size_t)row * DM;
    float s = 0.f;
    for (int i = tid; i < DM; i += 256) { float v = xr[i]; s += v * v; }
    __shared__ float red[256];
    red[tid] = s; __syncthreads();
    for (int off = 128; off > 0; off >>= 1) {
        if (tid < off) red[tid] += red[tid + off];
        __syncthreads();
    }
    float inv = rsqrtf(red[0] / (float)DM + 1e-5f);
    __bf16* orow = out + (size_t)row * DM;
    for (int i = tid; i < DM; i += 256) orow[i] = (__bf16)(xr[i] * inv * nw[i]);
}

// ---------------- fp32 -> bf16 casts ----------------
__global__ void k_cast(const float* __restrict__ src, __bf16* __restrict__ dst, long n) {
    long i = (long)blockIdx.x * 256 + threadIdx.x;
    if (i < n) dst[i] = (__bf16)src[i];
}

// strided + zero-padded cast: dst[r][c<valid]=src[r*srcStride+c], else 0
__global__ void k_cast_pad(const float* __restrict__ src, __bf16* __restrict__ dst,
                           int rows, int validCols, int srcStride, int dstCols) {
    long i = (long)blockIdx.x * 256 + threadIdx.x;
    long n = (long)rows * dstCols;
    if (i >= n) return;
    int r = (int)(i / dstCols), c = (int)(i % dstCols);
    dst[i] = (c < validCols) ? (__bf16)src[(long)r * srcStride + c] : (__bf16)0.0f;
}

// ---------------- WMMA bf16 GEMM:  C[M,N] = A[M,K] * W[N,K]^T ----------------
// 128x64 block tile, 8 waves (4M x 2N), wave -> 32x32 (4 WMMAs per K-step of 32).
// Double-buffered LDS filled by GLOBAL_LOAD_ASYNC_TO_LDS_B128 (ASYNCcnt pipeline).
// LDS layout per buffer (12 KB): As[128][32]bf16 @ +0, Bs[64][32]bf16 @ +8192.
// Requires: M % 128 == 0, K % 32 == 0 (true for all call sites); N handled by clamp+store-guard.
#define GEMM_LDS_BUF 12288
#define GEMM_LDS_TOT (2 * GEMM_LDS_BUF)
__global__ void __launch_bounds__(256)
k_gemm_bf16(const __bf16* __restrict__ A, const __bf16* __restrict__ W,
            float* __restrict__ C, int M, int N, int K) {
    extern __shared__ char smem[];

    int tid  = threadIdx.x;
    int bm   = blockIdx.y * 128;
    int bn   = blockIdx.x * 64;
    int wid  = tid >> 5, lane = tid & 31;
    int wm   = (wid & 3) * 32;          // M sub-tile (0,32,64,96)
    int wn   = (wid >> 2) * 32;         // N sub-tile (0,32)
    int fr   = lane & 15;               // fragment row within 16
    int fh   = lane >> 4;               // K half select

    // async-load mapping: As 4096 elems -> 2 x b128/thread; Bs 2048 elems -> 1 x b128/thread
    int ar = tid >> 1;                  // As row 0..127
    int ac = (tid & 1) * 16;            // As col {0,16}; chunks at ac, ac+8
    int br = tid >> 2;                  // Bs row 0..63
    int bc = (tid & 3) * 8;             // Bs col {0,8,16,24}

    int gmr = bm + ar; if (gmr >= M) gmr = M - 1;       // clamp (rows beyond N/M only feed
    int gnr = bn + br; if (gnr >= N) gnr = N - 1;       //  never-stored output columns)
    const __bf16* gA = A + (size_t)gmr * K + ac;
    const __bf16* gB = W + (size_t)gnr * K + bc;
    unsigned ldsA = (unsigned)((ar * 32 + ac) * 2);
    unsigned ldsB = (unsigned)(8192 + (br * 32 + bc) * 2);

    v8f c00 = {}, c01 = {}, c10 = {}, c11 = {};

    int nK = K >> 5;

    // prologue: tile 0 -> buffer 0
    gl_async_b128(ldsA,      gA);
    gl_async_b128(ldsA + 16, gA + 8);
    gl_async_b128(ldsB,      gB);

    for (int kt = 0; kt < nK; kt++) {
        int buf = kt & 1;
        // current tile's async ops are the only outstanding ones
        wait_async0();
        __syncthreads();   // tile kt complete in LDS for all waves; buf^1 free to overwrite

        if (kt + 1 < nK) {
            int k0 = (kt + 1) << 5;
            unsigned bb = (unsigned)((buf ^ 1) * GEMM_LDS_BUF);
            gl_async_b128(bb + ldsA,      gA + k0);
            gl_async_b128(bb + ldsA + 16, gA + k0 + 8);
            gl_async_b128(bb + ldsB,      gB + k0);
        }

        const __bf16* Asb = (const __bf16*)(smem + buf * GEMM_LDS_BUF);
        const __bf16* Bsb = (const __bf16*)(smem + buf * GEMM_LDS_BUF + 8192);

        v16bf a0 = *(const v16bf*)(Asb + (wm + fr) * 32 + fh * 16);
        v16bf a1 = *(const v16bf*)(Asb + (wm + 16 + fr) * 32 + fh * 16);
        v16bf b0 = *(const v16bf*)(Bsb + (wn + fr) * 32 + fh * 16);
        v16bf b1 = *(const v16bf*)(Bsb + (wn + 16 + fr) * 32 + fh * 16);

        c00 = __builtin_amdgcn_wmma_f32_16x16x32_bf16(false, a0, false, b0, (short)0, c00, false, false);
        c01 = __builtin_amdgcn_wmma_f32_16x16x32_bf16(false, a0, false, b1, (short)0, c01, false, false);
        c10 = __builtin_amdgcn_wmma_f32_16x16x32_bf16(false, a1, false, b0, (short)0, c10, false, false);
        c11 = __builtin_amdgcn_wmma_f32_16x16x32_bf16(false, a1, false, b1, (short)0, c11, false, false);
    }

    // C/D layout: lanes 0-15 -> rows vgpr i (0..7), lanes 16-31 -> rows 8+i; N = lane&15
    int cn  = bn + wn + fr;
    int cmb = bm + wm + fh * 8;
#pragma unroll
    for (int i = 0; i < 8; i++) {
        int cm0 = cmb + i;
        int cm1 = cmb + 16 + i;
        if (cm0 < M) {
            if (cn < N)      C[(size_t)cm0 * N + cn]      = c00[i];
            if (cn + 16 < N) C[(size_t)cm0 * N + cn + 16] = c01[i];
        }
        if (cm1 < M) {
            if (cn < N)      C[(size_t)cm1 * N + cn]      = c10[i];
            if (cn + 16 < N) C[(size_t)cm1 * N + cn + 16] = c11[i];
        }
    }
}

// ---------------- depthwise causal conv (width 4) + SiLU ----------------
__global__ void k_conv_silu(const float* __restrict__ xz, const float* __restrict__ cw,
                            const float* __restrict__ cb, float* __restrict__ xf,
                            __bf16* __restrict__ xb) {
    long i = (long)blockIdx.x * 256 + threadIdx.x;
    if (i >= (long)NROW * DI) return;
    int d = (int)(i % DI);
    long row = i / DI;
    int l = (int)(row % L_);
    long b = row / L_;
    float acc = cb[d];
#pragma unroll
    for (int k = 0; k < DC; k++) {
        int ll = l + k - (DC - 1);
        if (ll >= 0) acc += cw[d * DC + k] * xz[((long)b * L_ + ll) * (2 * DI) + d];
    }
    float s = siluf_(acc);
    xf[i] = s;
    xb[i] = (__bf16)s;
}

// ---------------- softplus(delta + bias), in place ----------------
__global__ void k_softplus(float* __restrict__ dl, const float* __restrict__ bias) {
    long i = (long)blockIdx.x * 256 + threadIdx.x;
    if (i >= (long)NROW * DI) return;
    float v = dl[i] + bias[i % DI];
    dl[i] = (v > 20.f) ? v : log1pf(__expf(v));
}

// ---------------- selective scan: one thread per (b, d), 16 states in registers ----------------
__global__ void __launch_bounds__(256)
k_scan(const float* __restrict__ delta, const float* __restrict__ xs,
       const float* __restrict__ dbc, const float* __restrict__ A_log,
       float* __restrict__ ys) {
    int t = blockIdx.x * 256 + threadIdx.x;
    if (t >= B_ * DI) return;
    int b = t / DI, d = t % DI;
    float As[DS], h[DS];
#pragma unroll
    for (int s = 0; s < DS; s++) { As[s] = -__expf(A_log[d * DS + s]); h[s] = 0.f; }
    for (int l = 0; l < L_; l++) {
        long row = (long)b * L_ + l;
        // hide HBM latency of the serial walk: prefetch ~8 rows ahead (global_prefetch_b8)
        long pr = row + 8; if (pr > (long)NROW - 1) pr = (long)NROW - 1;
        __builtin_prefetch(&delta[pr * DI + d], 0, 1);
        __builtin_prefetch(&xs[pr * DI + d], 0, 1);

        float dl = delta[row * DI + d];
        float xl = xs[row * DI + d];
        float dx = dl * xl;
        const float* bcp = dbc + row * (DTR + 2 * DS);   // Bm at +48, Cm at +64
        float y = 0.f;
#pragma unroll
        for (int s = 0; s < DS; s++) {
            float dA = __expf(dl * As[s]);
            h[s] = dA * h[s] + dx * bcp[DTR + s];
            y += h[s] * bcp[DTR + DS + s];
        }
        ys[row * DI + d] = y;
    }
}

// ---------------- gate: (ys + D*x) * silu(z) -> bf16 ----------------
__global__ void k_gate(const float* __restrict__ ys, const float* __restrict__ xs,
                       const float* __restrict__ xz, const float* __restrict__ Dp,
                       __bf16* __restrict__ yg) {
    long i = (long)blockIdx.x * 256 + threadIdx.x;
    if (i >= (long)NROW * DI) return;
    int d = (int)(i % DI);
    long row = i / DI;
    float z = xz[row * (2 * DI) + DI + d];
    yg[i] = (__bf16)((ys[i] + Dp[d] * xs[i]) * siluf_(z));
}

// ---------------- mean over L of (input + out) ----------------
__global__ void k_mean(const float* __restrict__ inp, const float* __restrict__ out,
                       float* __restrict__ e) {
    int t = blockIdx.x * 256 + threadIdx.x;
    if (t >= B_ * DM) return;
    int b = t / DM, dm = t % DM;
    float s = 0.f;
    for (int l = 0; l < L_; l++) {
        long idx = ((long)b * L_ + l) * DM + dm;
        s += inp[idx] + out[idx];
    }
    e[t] = s / (float)L_;
}

// ---------------- head FC1: xh = silu(e * W^T + b) ----------------
__global__ void k_fc1(const float* __restrict__ e, const float* __restrict__ w,
                      const float* __restrict__ bias, float* __restrict__ xh) {
    int t = blockIdx.x * 256 + threadIdx.x;
    if (t >= B_ * DIMIN) return;
    int b = t / DIMIN, i = t % DIMIN;
    const float* er = e + (long)b * DM;
    const float* wr = w + (long)i * DM;
    float s = bias[i];
    for (int k = 0; k < DM; k++) s += er[k] * wr[k];
    xh[t] = siluf_(s);
}

// ---------------- head FC2 + Poincare exp-map scaling ----------------
__global__ void __launch_bounds__(128)
k_fc2(const float* __restrict__ xh, const float* __restrict__ w,
      const float* __restrict__ bias, const float* __restrict__ scale,
      float* __restrict__ mu) {
    int b = blockIdx.x;
    int j = threadIdx.x;
    const float* xr = xh + (long)b * DIMIN;
    const float* wr = w + (long)j * DIMIN;
    float s = bias[j];
    for (int k = 0; k < DIMIN; k++) s += xr[k] * wr[k];
    s *= scale[0];
    __shared__ float red[128];
    red[j] = s * s; __syncthreads();
    for (int off = 64; off > 0; off >>= 1) {
        if (j < off) red[j] += red[j + off];
        __syncthreads();
    }
    float norm = fmaxf(sqrtf(red[0]), 1e-8f);
    const float sk = 0.5f;   // sqrt(0.25)
    float f = tanhf(sk * norm) / (sk * norm);
    mu[b * DIMOUT + j] = f * s;
}

// ==================== host launcher ====================
extern "C" void kernel_launch(void* const* d_in, const int* in_sizes, int n_in,
                              void* d_out, int out_size, void* d_ws, size_t ws_size,
                              hipStream_t stream) {
    (void)in_sizes; (void)n_in; (void)out_size; (void)ws_size;

    const float* input     = (const float*)d_in[0];
    const float* in_proj_w = (const float*)d_in[1];
    const float* conv_w    = (const float*)d_in[2];
    const float* conv_b    = (const float*)d_in[3];
    const float* x_proj_w  = (const float*)d_in[4];
    const float* dt_proj_w = (const float*)d_in[5];
    const float* dt_proj_b = (const float*)d_in[6];
    const float* A_log     = (const float*)d_in[7];
    const float* D_param   = (const float*)d_in[8];
    const float* out_proj_w= (const float*)d_in[9];
    const float* norm_w    = (const float*)d_in[10];
    const float* out_fc_w  = (const float*)d_in[11];
    const float* out_fc_b  = (const float*)d_in[12];
    const float* mu_fc_w   = (const float*)d_in[13];
    const float* mu_fc_b   = (const float*)d_in[14];
    const float* scale     = (const float*)d_in[15];

    char* ws = (char*)d_ws;
    size_t off = 0;
    auto alloc = [&](size_t bytes) -> char* {
        char* p = ws + off;
        off += (bytes + 255) & ~(size_t)255;
        return p;
    };

    __bf16* xn    = (__bf16*)alloc((size_t)NROW * DM * 2);
    __bf16* w_in  = (__bf16*)alloc((size_t)2 * DI * DM * 2);
    __bf16* w_xp  = (__bf16*)alloc((size_t)(DTR + 2 * DS) * DI * 2);
    __bf16* w_dt  = (__bf16*)alloc((size_t)DI * DTRP * 2);
    __bf16* w_out = (__bf16*)alloc((size_t)DM * DI * 2);
    float*  xz    = (float*)alloc((size_t)NROW * 2 * DI * 4);
    float*  xsf   = (float*)alloc((size_t)NROW * DI * 4);
    __bf16* xsb   = (__bf16*)alloc((size_t)NROW * DI * 2);
    float*  dbc   = (float*)alloc((size_t)NROW * (DTR + 2 * DS) * 4);
    __bf16* dt_bf = (__bf16*)alloc((size_t)NROW * DTRP * 2);
    float*  delta = (float*)alloc((size_t)NROW * DI * 4);
    float*  ysbuf = (float*)alloc((size_t)NROW * DI * 4);
    __bf16* yg    = (__bf16*)alloc((size_t)NROW * DI * 2);
    float*  outb  = (float*)alloc((size_t)NROW * DM * 4);
    float*  evec  = (float*)alloc((size_t)B_ * DM * 4);

    long n;

    // 1) RMSNorm -> bf16 activations
    k_rmsnorm<<<NROW, 256, 0, stream>>>(input, norm_w, xn);

    // 2) weight casts
    n = (long)2 * DI * DM;
    k_cast<<<(unsigned)((n + 255) / 256), 256, 0, stream>>>(in_proj_w, w_in, n);
    n = (long)(DTR + 2 * DS) * DI;
    k_cast<<<(unsigned)((n + 255) / 256), 256, 0, stream>>>(x_proj_w, w_xp, n);
    n = (long)DI * DTRP;
    k_cast_pad<<<(unsigned)((n + 255) / 256), 256, 0, stream>>>(dt_proj_w, w_dt, DI, DTR, DTR, DTRP);
    n = (long)DM * DI;
    k_cast<<<(unsigned)((n + 255) / 256), 256, 0, stream>>>(out_proj_w, w_out, n);

    // 3) in_proj GEMM: xz[8192,3072] = xn * w_in^T
    {
        dim3 g((2 * DI + 63) / 64, (NROW + 127) / 128);
        k_gemm_bf16<<<g, 256, GEMM_LDS_TOT, stream>>>(xn, w_in, xz, NROW, 2 * DI, DM);
    }

    // 4) conv + silu
    n = (long)NROW * DI;
    k_conv_silu<<<(unsigned)((n + 255) / 256), 256, 0, stream>>>(xz, conv_w, conv_b, xsf, xsb);

    // 5) x_proj GEMM: dbc[8192,80]
    {
        dim3 g((DTR + 2 * DS + 63) / 64, (NROW + 127) / 128);
        k_gemm_bf16<<<g, 256, GEMM_LDS_TOT, stream>>>(xsb, w_xp, dbc, NROW, DTR + 2 * DS, DI);
    }

    // 6) dt slice -> bf16 padded K=64
    n = (long)NROW * DTRP;
    k_cast_pad<<<(unsigned)((n + 255) / 256), 256, 0, stream>>>(dbc, dt_bf, NROW, DTR, DTR + 2 * DS, DTRP);

    // 7) dt_proj GEMM: delta[8192,1536]
    {
        dim3 g((DI + 63) / 64, (NROW + 127) / 128);
        k_gemm_bf16<<<g, 256, GEMM_LDS_TOT, stream>>>(dt_bf, w_dt, delta, NROW, DI, DTRP);
    }

    // 8) softplus(delta + bias)
    n = (long)NROW * DI;
    k_softplus<<<(unsigned)((n + 255) / 256), 256, 0, stream>>>(delta, dt_proj_b);

    // 9) selective scan
    k_scan<<<(B_ * DI + 255) / 256, 256, 0, stream>>>(delta, xsf, dbc, A_log, ysbuf);

    // 10) gate
    k_gate<<<(unsigned)((n + 255) / 256), 256, 0, stream>>>(ysbuf, xsf, xz, D_param, yg);

    // 11) out_proj GEMM: outb[8192,768]
    {
        dim3 g((DM + 63) / 64, (NROW + 127) / 128);
        k_gemm_bf16<<<g, 256, GEMM_LDS_TOT, stream>>>(yg, w_out, outb, NROW, DM, DI);
    }

    // 12) residual + mean over L
    k_mean<<<(B_ * DM + 255) / 256, 256, 0, stream>>>(input, outb, evec);

    // 13) heads -> d_out = [xh (8*768) | mu (8*128)]
    float* xh = (float*)d_out;
    float* mu = xh + B_ * DIMIN;
    k_fc1<<<(B_ * DIMIN + 255) / 256, 256, 0, stream>>>(evec, out_fc_w, out_fc_b, xh);
    k_fc2<<<B_, 128, 0, stream>>>(xh, mu_fc_w, mu_fc_b, scale, mu);
}
